// XORWrapper_7164005449756
// MI455X (gfx1250) — 8-bit matrix operand order repaired and checked by reading, compiled, not run on hardware
//
#include <hip/hip_runtime.h>
#include <hip/hip_bf16.h>


typedef signed char s8;
typedef int v8i __attribute__((ext_vector_type(8)));

#define OFF_ALPHA 0ul
#define OFF_XS    256ul
#define OFF_W1B   (OFF_XS + 4816896ul)
#define OFF_H1S   (OFF_W1B + 4096ul)
#define OFF_W2B   (OFF_H1S + 25690112ul)
#define OFF_H2S   (OFF_W2B + 73728ul)
#define OFF_W3B   (OFF_H2S + 12845056ul)
#define OFF_H3F   (OFF_W3B + 294912ul)
#define OFF_HPS   (OFF_H3F + 25690112ul)
#define OFF_WFS   (OFF_HPS + 8192ul)
#define OFF_ZERO  (OFF_WFS + 256000ul)

__global__ void abs_sums_kernel(const float* w1, int n1, const float* w2, int n2,
                                const float* w3, int n3, const float* wf, int nf,
                                float* out) {
  __shared__ float red[256];
  const float* w; int n;
  switch (blockIdx.x) {
    case 0: w = w1; n = n1; break;
    case 1: w = w2; n = n2; break;
    case 2: w = w3; n = n3; break;
    default: w = wf; n = nf; break;
  }
  float s = 0.f;
  for (int i = threadIdx.x; i < n; i += 256) s += fabsf(w[i]);
  red[threadIdx.x] = s;
  __syncthreads();
  for (int off = 128; off > 0; off >>= 1) {
    if (threadIdx.x < off) red[threadIdx.x] += red[threadIdx.x + off];
    __syncthreads();
  }
  if (threadIdx.x == 0) out[blockIdx.x] = red[0];
}

__global__ void bin_conv_w(const float* __restrict__ w, s8* __restrict__ out,
                           int Cin, int Kpad, int total) {
  int t = blockIdx.x * 256 + threadIdx.x;
  if (t >= total) return;
  int o = t / Kpad, k = t - o * Kpad;
  s8 v = 0;
  if (k < 9 * Cin) {
    int p = k / Cin, c = k - p * Cin;
    float f = w[(o * Cin + c) * 9 + p];
    v = (f >= 0.f) ? (s8)1 : (s8)-1;
  }
  out[t] = v;
}

__global__ void bin_fc_w(const float* __restrict__ wf, s8* __restrict__ out, int total) {
  int t = blockIdx.x * 256 + threadIdx.x;
  if (t >= total) return;
  out[t] = (wf[t] >= 0.f) ? (s8)1 : (s8)-1;
}

__global__ void bin_x(const float* __restrict__ x, s8* __restrict__ xs) {
  int t = blockIdx.x * 256 + threadIdx.x;
  int c = t % 3, r = t / 3;
  int wc = r % 224; r /= 224;
  int hr = r % 224; int n = r / 224;
  float f = x[((n * 3 + c) * 224 + hr) * 224 + wc];
  xs[t] = (f >= 0.f) ? (s8)1 : (s8)-1;
}

__global__ void zero_pad(int* z) { z[threadIdx.x] = 0; }

__global__ void conv1_direct(const s8* __restrict__ xs, const s8* __restrict__ w1b,
                             const float* __restrict__ b1, const float* __restrict__ alphas,
                             s8* __restrict__ h1s) {
  int t = blockIdx.x * 256 + threadIdx.x;
  int o = t & 63, pix = t >> 6;
  int n = pix / (112 * 112), rem = pix % (112 * 112);
  int oy = rem / 112, ox = rem % 112;
  const s8* wr = w1b + o * 64;
  int acc = 0;
#pragma unroll
  for (int p = 0; p < 9; p++) {
    int ky = p / 3, kx = p % 3;
    int iy = oy * 2 - 1 + ky, ix = ox * 2 - 1 + kx;
    if (iy >= 0 && iy < 224 && ix >= 0 && ix < 224) {
      const s8* xb = xs + ((n * 224 + iy) * 224 + ix) * 3;
      acc += (int)xb[0] * wr[p * 3 + 0] + (int)xb[1] * wr[p * 3 + 1] + (int)xb[2] * wr[p * 3 + 2];
    }
  }
  float a = alphas[0] * (1.f / 1728.f);
  float y = a * a * (float)acc + b1[o];
  h1s[t] = (y >= 0.f) ? (s8)1 : (s8)-1;
}

template <bool STORE_FLOAT>
__global__ void conv_wmma_iu8(const char* __restrict__ wsbase, int in_off, int zero_off,
                              const s8* __restrict__ wq,
                              const float* __restrict__ bias,
                              const float* __restrict__ alphas, int aidx, float inv_cnt,
                              void* __restrict__ outp,
                              int IH, int IW, int C, int OH, int OW, int Cout, int Kpad) {
  __shared__ __align__(16) int shm[2][1024];

  const int tid  = threadIdx.x;
  const int lane = tid & 31;
  const int wv   = tid >> 5;
  const int lgrp = lane >> 4;
  const int lmod = lane & 15;

  const int n_ch   = blockIdx.y * 128 + wv * 16 + lmod;
  const int m_base = blockIdx.x * 64;

  const float a   = alphas[aidx] * inv_cnt;
  const float asq = a * a;
  const float bch = bias[n_ch];

  const int srow  = tid >> 2;
  const int seg16 = (tid & 3) * 16;
  const int hw = OH * OW;
  {
  }
  int sm  = m_base + srow;
  int sni = sm / hw, srem = sm - sni * hw;
  int soy = srem / OW, sox = srem - soy * OW;
  const int siy0 = soy * 2 - 1, six0 = sox * 2 - 1;
  const int srowbase = (sni * IH) * IW;

  const unsigned lds0 = (unsigned)(size_t)(&shm[0][0]);

  const int ksteps = Kpad >> 6;
  const s8* wrow = wq + n_ch * Kpad;

  auto issue = [&](int kk) {
    int kabs0 = kk * 64;
    int p  = kabs0 / C;
    int cb = kabs0 - p * C;
    int ky = p / 3, kx = p - ky * 3;
    int iy = siy0 + ky, ix = six0 + kx;
    bool valid = (iy >= 0) & (iy < IH) & (ix >= 0) & (ix < IW);
    int goff = valid ? (in_off + ((sni * IH + iy) * IW + ix) * C + cb + seg16)
                     : (zero_off + seg16);
    unsigned ldst = lds0 + (unsigned)((kk & 1) * 4096 + srow * 64 + seg16);
    asm volatile("global_load_async_to_lds_b128 %0, %1, %2 offset:0"
                 :: "v"(ldst), "v"(goff), "s"(wsbase) : "memory");
  };

  v8i acc[4];
#pragma unroll
  for (int s = 0; s < 4; s++) {
#pragma unroll
    for (int i = 0; i < 8; i++) acc[s][i] = 0;
  }

  issue(0);
  for (int kk = 0; kk < ksteps; ++kk) {
    if (kk + 1 < ksteps) {
      issue(kk + 1);
      asm volatile("s_wait_asynccnt 0x1" ::: "memory");
    } else {
      asm volatile("s_wait_asynccnt 0x0" ::: "memory");
    }
    __syncthreads();

    if (kk + 1 < ksteps) __builtin_prefetch(wrow + (kk + 1) * 64, 0, 1);
    v8i B;
#pragma unroll
    for (int v = 0; v < 8; v++) {
      int kb = ((v & 1) * 4) + ((v >> 1) * 16) + lgrp * 8;
      B[v] = *reinterpret_cast<const int*>(wrow + kk * 64 + kb);
    }

    const int bufo = (kk & 1) * 1024;
#pragma unroll
    for (int s = 0; s < 4; s++) {
      int rbase = bufo + (s * 16 + lmod) * 16 + lgrp * 2;
      v8i A;
#pragma unroll
      for (int vp = 0; vp < 4; vp++) {
        int2 d = *reinterpret_cast<const int2*>(&shm[0][0] + rbase + vp * 4);
        A[2 * vp]     = d.x;
        A[2 * vp + 1] = d.y;
      }
      acc[s] = __builtin_amdgcn_wmma_i32_16x16x64_iu8(true, A, true, B, acc[s], false, false);
    }
    __syncthreads();
  }

#pragma unroll
  for (int s = 0; s < 4; s++) {
#pragma unroll
    for (int r = 0; r < 8; r++) {
      int m  = m_base + s * 16 + lgrp * 8 + r;
      float y = asq * (float)acc[s][r] + bch;
      if (STORE_FLOAT) {
        float yc = fminf(fmaxf(y, -1.f), 1.f);
        reinterpret_cast<float*>(outp)[m * Cout + n_ch] = yc;
      } else {
        reinterpret_cast<s8*>(outp)[m * Cout + n_ch] = (y >= 0.f) ? (s8)1 : (s8)-1;
      }
    }
  }
}

__global__ void pool_sign(const float* __restrict__ h3, s8* __restrict__ hps) {
  int n = blockIdx.x, c = threadIdx.x;
  float s = 0.f;
  for (int sp = 0; sp < 784; sp++) s += h3[(n * 784 + sp) * 256 + c];
  hps[n * 256 + c] = (s >= 0.f) ? (s8)1 : (s8)-1;
}

__global__ void fc_kernel(const s8* __restrict__ hps, const s8* __restrict__ wfs,
                          const float* __restrict__ bf, const float* __restrict__ alphas,
                          float* __restrict__ out) {
  int t = blockIdx.x * 256 + threadIdx.x;
  if (t >= 32000) return;
  int n = t / 1000, o = t - n * 1000;
  const s8* hp = hps + n * 256;
  const s8* wr = wfs + o * 256;
  int acc = 0;
#pragma unroll 4
  for (int k = 0; k < 256; k++) acc += (int)hp[k] * (int)wr[k];
  float a = alphas[3] * (1.f / 256000.f);
  out[t] = a * a * (float)acc + bf[o];
}

extern "C" void kernel_launch(void* const* d_in, const int* in_sizes, int n_in,
                              void* d_out, int out_size, void* d_ws, size_t ws_size,
                              hipStream_t stream) {
  const float* x  = (const float*)d_in[0];
  const float* w1 = (const float*)d_in[1];
  const float* b1 = (const float*)d_in[2];
  const float* w2 = (const float*)d_in[3];
  const float* b2 = (const float*)d_in[4];
  const float* w3 = (const float*)d_in[5];
  const float* b3 = (const float*)d_in[6];
  const float* wf = (const float*)d_in[7];
  const float* bf = (const float*)d_in[8];

  char* ws = (char*)d_ws;
  float* alphas = (float*)(ws + OFF_ALPHA);
  s8*    xs  = (s8*)(ws + OFF_XS);
  s8*    w1b = (s8*)(ws + OFF_W1B);
  s8*    h1s = (s8*)(ws + OFF_H1S);
  s8*    w2b = (s8*)(ws + OFF_W2B);
  s8*    h2s = (s8*)(ws + OFF_H2S);
  s8*    w3b = (s8*)(ws + OFF_W3B);
  float* h3f = (float*)(ws + OFF_H3F);
  s8*    hps = (s8*)(ws + OFF_HPS);
  s8*    wfs = (s8*)(ws + OFF_WFS);
  float* out = (float*)d_out;

  abs_sums_kernel<<<4, 256, 0, stream>>>(w1, 1728, w2, 73728, w3, 294912, wf, 256000, alphas);

  bin_conv_w<<<(64 * 64) / 256, 256, 0, stream>>>(w1, w1b, 3, 64, 64 * 64);
  bin_conv_w<<<(128 * 576) / 256, 256, 0, stream>>>(w2, w2b, 64, 576, 128 * 576);
  bin_conv_w<<<(256 * 1152) / 256, 256, 0, stream>>>(w3, w3b, 128, 1152, 256 * 1152);
  bin_fc_w<<<256000 / 256, 256, 0, stream>>>(wf, wfs, 256000);
  bin_x<<<(32 * 224 * 224 * 3) / 256, 256, 0, stream>>>(x, xs);
  zero_pad<<<1, 64, 0, stream>>>((int*)(ws + OFF_ZERO));

  conv1_direct<<<(32 * 112 * 112 * 64) / 256, 256, 0, stream>>>(xs, w1b, b1, alphas, h1s);

  conv_wmma_iu8<false><<<dim3(100352 / 64, 1), 256, 0, stream>>>(
      ws, (int)OFF_H1S, (int)OFF_ZERO, w2b, b2, alphas, 1, 1.f / 73728.f, (void*)h2s,
      112, 112, 64, 56, 56, 128, 576);

  conv_wmma_iu8<true><<<dim3(25088 / 64, 2), 256, 0, stream>>>(
      ws, (int)OFF_H2S, (int)OFF_ZERO, w3b, b3, alphas, 2, 1.f / 294912.f, (void*)h3f,
      56, 56, 128, 28, 28, 256, 1152);

  pool_sign<<<32, 256, 0, stream>>>(h3f, hps);
  fc_kernel<<<(32000 + 255) / 256, 256, 0, stream>>>(hps, wfs, bf, alphas, out);
}
